// AQT_skt_model_66666482369144
// MI455X (gfx1250) — compile-verified, matrix operand order checked by reading, not run on hardware
//
#include <hip/hip_runtime.h>
#include <hip/hip_bf16.h>

// ---------------------------------------------------------------------------
// CDNA5 (gfx1250) WMMA transformer forward pass, round 4.
// - All GEMMs: v_wmma_f32_16x16x32_f16, f32 accumulate.
// - Weights prepacked to f16 fragment-order (zero-padded K/N).
// - GEMM blocks stage their (contiguous) packed-B span into LDS with the
//   Tensor Data Mover (tensor_load_to_lds + s_wait_tensorcnt, 6-arg builtin),
//   then read B fragments via ds_load_b128.
// ---------------------------------------------------------------------------

typedef __attribute__((ext_vector_type(16))) _Float16 v16h;
typedef __attribute__((ext_vector_type(8)))  float    v8f;
typedef __attribute__((ext_vector_type(4)))  unsigned v4u;
typedef __attribute__((ext_vector_type(8)))  int      v8i;
typedef __attribute__((ext_vector_type(4)))  int      v4i;

#define BS     256
#define DMODEL 128
#define NH     4
#define HD     32
#define NP     196
#define PD     768
#define NA     15
#define GEMM_MAX_KT 8   // max Ktiles for the LDS-staged GEMM (32 KB stage)

__device__ __forceinline__ int frag_k(int i, int khalf) {
  return (i < 4) ? (khalf + 2 * i) : (16 + khalf + 2 * (i - 4));
}

__device__ __forceinline__ v8f wmma_f16(v16h a, v16h b, v8f c) {
  return __builtin_amdgcn_wmma_f32_16x16x32_f16(false, a, false, b, (short)0, c,
                                                false, false);
}

__device__ __forceinline__ v16h pack16(float4 x0, float4 x1, float4 x2,
                                       float4 x3) {
  v16h a;
  a[0] = (_Float16)x0.x;  a[1] = (_Float16)x0.y;
  a[2] = (_Float16)x0.z;  a[3] = (_Float16)x0.w;
  a[4] = (_Float16)x1.x;  a[5] = (_Float16)x1.y;
  a[6] = (_Float16)x1.z;  a[7] = (_Float16)x1.w;
  a[8] = (_Float16)x2.x;  a[9] = (_Float16)x2.y;
  a[10] = (_Float16)x2.z; a[11] = (_Float16)x2.w;
  a[12] = (_Float16)x3.x; a[13] = (_Float16)x3.y;
  a[14] = (_Float16)x3.z; a[15] = (_Float16)x3.w;
  return a;
}

__device__ __forceinline__ v16h load_a_vec(const float4* __restrict__ Ar,
                                           int kt, int k4) {
  int q = kt * 8;
  return pack16(Ar[q + k4], Ar[q + k4 + 1], Ar[q + 4 + k4], Ar[q + 5 + k4]);
}

// ---------------------------------------------------------------------------
// Weight prepack: f32 KxN (row-major) -> f16 fragment-order tiles,
// zero-padded to Ktiles*32 x NtPad*16.  Layout: ((nt*Ktiles+kt)*32+lane)*16.
// ---------------------------------------------------------------------------
__global__ void prepack_weight(const float* __restrict__ B, int ldb, int K,
                               int N, _Float16* __restrict__ out, int Ktiles,
                               int NtPad) {
  int idx = blockIdx.x * blockDim.x + threadIdx.x;
  int total = NtPad * Ktiles * 32;
  if (idx >= total) return;
  int lane = idx & 31;
  int kt = (idx >> 5) % Ktiles;
  int nt = (idx >> 5) / Ktiles;
  int col = nt * 16 + (lane & 15);
  int khalf = (lane >> 4) ? 8 : 0;
  int k0 = kt * 32;
  bool cok = (col < N);
  v16h b;
#pragma unroll
  for (int i = 0; i < 8; ++i) {
    int k = k0 + frag_k(i, khalf);
    float x0 = (cok && k < K) ? B[(size_t)k * ldb + col] : 0.0f;
    float x1 = (cok && k + 1 < K) ? B[(size_t)(k + 1) * ldb + col] : 0.0f;
    b[2 * i] = (_Float16)x0;
    b[2 * i + 1] = (_Float16)x1;
  }
  *(v16h*)(out + (size_t)idx * 16) = b;
}

// ---------------------------------------------------------------------------
// GEMM: C(MxN) = act(A @ Bpacked + bias).  Wave = 16 rows x 64 cols.
// block = 128 (4 waves); grid = (M/64, NtPad/4).  Ktiles <= GEMM_MAX_KT.
// The block's 4 N-tiles of packed B are one contiguous span -> TDM to LDS.
// ---------------------------------------------------------------------------
__global__ void gemm_wmma(const float* __restrict__ A, int lda,
                          const _Float16* __restrict__ Bp, int Ktiles,
                          const float* __restrict__ bias,
                          float* __restrict__ C, int ldc,
                          int M, int N, int relu) {
  __shared__ __align__(32) _Float16 sB[GEMM_MAX_KT * 4 * 32 * 16];  // 32 KB
  int wave = threadIdx.x >> 5;
  int lane = threadIdx.x & 31;
  int nt0 = blockIdx.y * 4;
  const _Float16* Bblk = Bp + (size_t)nt0 * Ktiles * 512;
  unsigned nhalf = (unsigned)(4 * Ktiles * 512);  // <= 16384

#if defined(__has_builtin) && __has_builtin(__builtin_amdgcn_tensor_load_to_lds)
  if (wave == 0) {
    unsigned ldsaddr = (unsigned)(size_t)(void*)&sB[0];
    unsigned long long ga = (unsigned long long)(size_t)Bblk;
    // D# group 0: count=1 | lds_addr | global_addr[56:0] | type=2
    v4u g0;
    g0[0] = 1u;
    g0[1] = ldsaddr;
    g0[2] = (unsigned)(ga & 0xffffffffu);
    g0[3] = ((unsigned)((ga >> 32) & 0x01ffffffu)) | 0x80000000u;
    // D# group 1: 1-D copy of nhalf 2-byte elements.
    v8i g1;
    g1[0] = (int)(1u << 16);                       // data_size = 2 bytes
    g1[1] = (int)((nhalf & 0xffffu) << 16);        // tensor_dim0[15:0]
    g1[2] = (int)(((nhalf >> 16) & 0xffffu) | (1u << 16));  // dim0 hi | dim1=1
    g1[3] = (int)(nhalf << 16);                    // tile_dim0 = nhalf
    g1[4] = 0;                                     // tile_dim1/2 unused
    g1[5] = (int)nhalf;                            // tensor_dim0_stride
    g1[6] = 0;
    g1[7] = 0;
    v4i zz = {};
    v8i z8 = {};
    __builtin_amdgcn_tensor_load_to_lds(g0, g1, zz, zz, z8, 0);
    __builtin_amdgcn_s_wait_tensorcnt(0);
  }
  __syncthreads();
#else
  {
    const float4* gsrc = (const float4*)Bblk;
    float4* ldst = (float4*)sB;
    int nchunks = (int)(nhalf >> 3);  // 16 B per chunk
    for (int i = threadIdx.x; i < nchunks; i += 128) ldst[i] = gsrc[i];
  }
  __syncthreads();
#endif

  int rt = blockIdx.x * 4 + wave;
  int row0 = rt * 16;
  if (row0 < M) {
    int k4 = (lane >> 4) ? 2 : 0;
    const float4* Ar = (const float4*)(A + (size_t)(row0 + (lane & 15)) * lda);
    __builtin_prefetch(Ar, 0, 0);
    const v16h* Bv = (const v16h*)sB;
    v8f acc[4] = {};
#pragma unroll 1
    for (int kt = 0; kt < Ktiles; ++kt) {
      v16h a = load_a_vec(Ar, kt, k4);
#pragma unroll
      for (int j = 0; j < 4; ++j) {
        v16h b = Bv[((size_t)j * Ktiles + kt) * 32 + lane];
        acc[j] = wmma_f16(a, b, acc[j]);
      }
    }
    int rbase = row0 + ((lane >> 4) ? 8 : 0);
#pragma unroll
    for (int j = 0; j < 4; ++j) {
      int col = (nt0 + j) * 16 + (lane & 15);
      if (col < N) {
        float bv = bias ? bias[col] : 0.0f;
#pragma unroll
        for (int r = 0; r < 8; ++r) {
          float v = acc[j][r] + bv;
          if (relu) v = fmaxf(v, 0.0f);
          C[(size_t)(rbase + r) * ldc + col] = v;
        }
      }
    }
  }
}

// ---------------------------------------------------------------------------
// Patch embedding (gather fused into A fragments, packed patch_W for B).
// Ktiles = 24 (> stage capacity) -> direct global B loads.
// ---------------------------------------------------------------------------
__global__ void patch_embed_kernel(const float* __restrict__ img,
                                   const _Float16* __restrict__ Wp,
                                   const float* __restrict__ bp,
                                   const float* __restrict__ pos,
                                   float* __restrict__ out) {
  int wave = threadIdx.x >> 5;
  int lane = threadIdx.x & 31;
  int rt = blockIdx.x * 4 + wave;
  int row0 = rt * 16;
  int nt0 = blockIdx.y * 4;
  int t = row0 + (lane & 15);
  int n = t >> 8;
  int bb_ = t & 255;
  const float* imgb = img + (size_t)bb_ * 150528 +
                      (size_t)(n / 14) * (16 * 224) + (size_t)(n % 14) * 16;
  int khalf = (lane >> 4) ? 8 : 0;
  const v16h* Bv = (const v16h*)Wp;
  const int Ktiles = PD / 32;  // 24
  v8f acc[4] = {};
#pragma unroll 1
  for (int kt = 0; kt < Ktiles; ++kt) {
    int k0 = kt * 32;
    v16h a;
#pragma unroll
    for (int i = 0; i < 8; ++i) {
      int k = k0 + frag_k(i, khalf);
      int c0 = k % 3, pp0 = k / 3;
      int c1 = (k + 1) % 3, pp1 = (k + 1) / 3;
      a[2 * i] =
          (_Float16)imgb[(size_t)c0 * 50176 + (pp0 >> 4) * 224 + (pp0 & 15)];
      a[2 * i + 1] =
          (_Float16)imgb[(size_t)c1 * 50176 + (pp1 >> 4) * 224 + (pp1 & 15)];
    }
#pragma unroll
    for (int j = 0; j < 4; ++j) {
      v16h b = Bv[((size_t)(nt0 + j) * Ktiles + kt) * 32 + lane];
      acc[j] = wmma_f16(a, b, acc[j]);
    }
  }
  int rbase = row0 + ((lane >> 4) ? 8 : 0);
#pragma unroll
  for (int j = 0; j < 4; ++j) {
    int col = (nt0 + j) * 16 + (lane & 15);
    float bv = bp[col];
#pragma unroll
    for (int r = 0; r < 8; ++r) {
      int tt = rbase + r;
      int nn = tt >> 8;
      out[(size_t)tt * DMODEL + col] = acc[j][r] + bv + pos[nn * DMODEL + col];
    }
  }
}

// ---------------------------------------------------------------------------
// Attention scores: one 16x16x32 WMMA per tile (hd == 32).  block = 32.
// ---------------------------------------------------------------------------
__global__ void attn_scores(const float* __restrict__ Q, int ldq, int qoff,
                            const float* __restrict__ Kp, int ldk, int koff,
                            float* __restrict__ S, int ldS, int lqpad,
                            int Lq, int Lk, int bh_base, float scale) {
  int lane = threadIdx.x & 31;
  int bh = bh_base + blockIdx.z;
  int b = bh >> 2, h = bh & 3;
  int lt = blockIdx.x, nt = blockIdx.y;
  int l = lt * 16 + (lane & 15); if (l >= Lq) l = Lq - 1;
  int n = nt * 16 + (lane & 15); if (n >= Lk) n = Lk - 1;
  int k4 = (lane >> 4) ? 2 : 0;
  const float4* q4 =
      (const float4*)(Q + (size_t)(l * BS + b) * ldq + qoff + h * HD);
  const float4* kk4 =
      (const float4*)(Kp + (size_t)(n * BS + b) * ldk + koff + h * HD);
  v16h a = pack16(q4[k4], q4[k4 + 1], q4[4 + k4], q4[5 + k4]);
  v16h bf = pack16(kk4[k4], kk4[k4 + 1], kk4[4 + k4], kk4[5 + k4]);
  v8f c = {};
  c = wmma_f16(a, bf, c);
  float* Sz = S + (size_t)blockIdx.z * lqpad * ldS;
  int row0 = lt * 16 + ((lane >> 4) ? 8 : 0);
  int col = nt * 16 + (lane & 15);
  if (col < Lk) {
#pragma unroll
    for (int r = 0; r < 8; ++r)
      Sz[(size_t)(row0 + r) * ldS + col] = c[r] * scale;
  }
}

// Row softmax over Lk; zero-fill [Lk, ldS).  One wave per row.
__global__ void softmax_rows(float* __restrict__ S, int ldS, int lqpad, int Lq,
                             int Lk) {
  int wave = threadIdx.x >> 5, lane = threadIdx.x & 31;
  int row = blockIdx.x * 4 + wave;
  int z = row / Lq, l = row % Lq;
  float* p = S + (size_t)(z * lqpad + l) * ldS;
  float mx = -1e30f;
  for (int i = lane; i < Lk; i += 32) mx = fmaxf(mx, p[i]);
  for (int m = 16; m; m >>= 1) mx = fmaxf(mx, __shfl_xor(mx, m, 32));
  float sum = 0.0f;
  for (int i = lane; i < Lk; i += 32) {
    float e = __expf(p[i] - mx);
    p[i] = e;
    sum += e;
  }
  for (int m = 16; m; m >>= 1) sum += __shfl_xor(sum, m, 32);
  float inv = 1.0f / sum;
  for (int i = lane; i < Lk; i += 32) p[i] *= inv;
  for (int i = Lk + lane; i < ldS; i += 32) p[i] = 0.0f;
}

// O = P @ V per (b,h).  block = 32; grid = (lqpad/16, 2, nbh_chunk).
__global__ void attn_out(const float* __restrict__ S, int ldS, int lqpad,
                         const float* __restrict__ V, int ldv, int voff,
                         float* __restrict__ O, int ldo, int ooff,
                         int Lq, int Lk, int bh_base) {
  int lane = threadIdx.x & 31;
  int bh = bh_base + blockIdx.z;
  int b = bh >> 2, h = bh & 3;
  int lt = blockIdx.x, ct = blockIdx.y;
  int l = lt * 16 + (lane & 15);
  int k4 = (lane >> 4) ? 2 : 0;
  int khalf = k4 * 4;
  int nc = lane & 15;
  const float4* srow =
      (const float4*)(S + ((size_t)blockIdx.z * lqpad + l) * ldS);
  const float* vcol = V + (size_t)b * ldv + voff + h * HD + ct * 16 + nc;
  v8f c = {};
  int full = ldS & ~31;
#pragma unroll 1
  for (int k0 = 0; k0 < full; k0 += 32) {
    int q = k0 >> 2;
    v16h a = pack16(srow[q + k4], srow[q + k4 + 1], srow[q + 4 + k4],
                    srow[q + 5 + k4]);
    v16h bf;
#pragma unroll
    for (int i = 0; i < 8; ++i) {
      int k = k0 + frag_k(i, khalf);
      int ka = (k < Lk) ? k : (Lk - 1);
      int kb = (k + 1 < Lk) ? (k + 1) : (Lk - 1);
      bf[2 * i] = (_Float16)vcol[(size_t)ka * BS * ldv];
      bf[2 * i + 1] = (_Float16)vcol[(size_t)kb * BS * ldv];
    }
    c = wmma_f16(a, bf, c);
  }
  if (ldS & 16) {
    int k0 = full;
    int q = k0 >> 2;
    float4 x0 = srow[q + k4], x1 = srow[q + k4 + 1];
    v16h a, bf;
    a[0] = (_Float16)x0.x; a[1] = (_Float16)x0.y;
    a[2] = (_Float16)x0.z; a[3] = (_Float16)x0.w;
    a[4] = (_Float16)x1.x; a[5] = (_Float16)x1.y;
    a[6] = (_Float16)x1.z; a[7] = (_Float16)x1.w;
#pragma unroll
    for (int i = 8; i < 16; ++i) a[i] = (_Float16)0.0f;
#pragma unroll
    for (int i = 0; i < 4; ++i) {
      int k = k0 + khalf + 2 * i;
      int ka = (k < Lk) ? k : (Lk - 1);
      int kb = (k + 1 < Lk) ? (k + 1) : (Lk - 1);
      bf[2 * i] = (_Float16)vcol[(size_t)ka * BS * ldv];
      bf[2 * i + 1] = (_Float16)vcol[(size_t)kb * BS * ldv];
    }
#pragma unroll
    for (int i = 8; i < 16; ++i) bf[i] = (_Float16)0.0f;
    c = wmma_f16(a, bf, c);
  }
  int row0 = lt * 16 + ((lane >> 4) ? 8 : 0);
  int col = ct * 16 + nc;
#pragma unroll
  for (int r = 0; r < 8; ++r) {
    int row = row0 + r;
    if (row < Lq)
      O[(size_t)(row * BS + b) * ldo + ooff + h * HD + col] = c[r];
  }
}

// LayerNorm over D=128 with optional residual; one wave per row.
__global__ void layernorm_k(const float* __restrict__ x,
                            const float* __restrict__ res,
                            const float* __restrict__ s,
                            const float* __restrict__ bb,
                            float* __restrict__ out, int nrows) {
  int wave = threadIdx.x >> 5, lane = threadIdx.x & 31;
  int row = blockIdx.x * 4 + wave;
  if (row >= nrows) return;
  const float* xr = x + (size_t)row * DMODEL;
  const float* rr = res ? res + (size_t)row * DMODEL : nullptr;
  float v[4];
  float m = 0.0f;
#pragma unroll
  for (int j = 0; j < 4; ++j) {
    int i = lane + 32 * j;
    v[j] = xr[i] + (rr ? rr[i] : 0.0f);
    m += v[j];
  }
  for (int t = 16; t; t >>= 1) m += __shfl_xor(m, t, 32);
  m *= (1.0f / 128.0f);
  float var = 0.0f;
#pragma unroll
  for (int j = 0; j < 4; ++j) {
    float d = v[j] - m;
    var += d * d;
  }
  for (int t = 16; t; t >>= 1) var += __shfl_xor(var, t, 32);
  var *= (1.0f / 128.0f);
  float inv = rsqrtf(var + 1e-5f);
#pragma unroll
  for (int j = 0; j < 4; ++j) {
    int i = lane + 32 * j;
    out[(size_t)row * DMODEL + i] = (v[j] - m) * inv * s[i] + bb[i];
  }
}

// sk (256x200) -> zero-padded (256x224).
__global__ void pad_sk(const float* __restrict__ sk, float* __restrict__ out) {
  int idx = blockIdx.x * blockDim.x + threadIdx.x;
  if (idx >= 256 * 224) return;
  int r = idx / 224, c = idx % 224;
  out[idx] = (c < 200) ? sk[r * 200 + c] : 0.0f;
}

// feats[(a*256+b), 0..127]: [one-hot(15)|angle,px,py|s3(100)|zeros(10)]
__global__ void build_feats(const float* __restrict__ angle,
                            const float* __restrict__ px,
                            const float* __restrict__ py,
                            const float* __restrict__ s3,
                            float* __restrict__ feats) {
  int idx = blockIdx.x * blockDim.x + threadIdx.x;
  if (idx >= NA * BS * 128) return;
  int t = idx >> 7, col = idx & 127;
  int a = t >> 8, b = t & 255;
  float v = 0.0f;
  if (col < 15)        v = (col == a) ? 1.0f : 0.0f;
  else if (col == 15)  v = angle[b];
  else if (col == 16)  v = px[b];
  else if (col == 17)  v = py[b];
  else if (col < 118)  v = s3[b * 100 + (col - 18)];
  feats[idx] = v;
}

// q[b, a] = dot(h1[(a*256+b), :], l2_W) + l2_b.  One wave per row.
__global__ void head_final(const float* __restrict__ h1,
                           const float* __restrict__ w,
                           const float* __restrict__ bptr,
                           float* __restrict__ out) {
  int wave = threadIdx.x >> 5, lane = threadIdx.x & 31;
  int t = blockIdx.x * 4 + wave;
  if (t >= NA * BS) return;
  float sum = 0.0f;
#pragma unroll
  for (int j = 0; j < 4; ++j) {
    int i = lane + 32 * j;
    sum += h1[(size_t)t * DMODEL + i] * w[i];
  }
  for (int m = 16; m; m >>= 1) sum += __shfl_xor(sum, m, 32);
  if (lane == 0) {
    int a = t >> 8, b = t & 255;
    out[b * NA + a] = sum + bptr[0];
  }
}

// ---------------------------------------------------------------------------
extern "C" void kernel_launch(void* const* d_in, const int* in_sizes, int n_in,
                              void* d_out, int out_size, void* d_ws,
                              size_t ws_size, hipStream_t stream) {
  const float* image     = (const float*)d_in[0];
  const float* angle     = (const float*)d_in[1];
  const float* pos_x     = (const float*)d_in[2];
  const float* pos_y     = (const float*)d_in[3];
  const float* sk        = (const float*)d_in[4];
  const float* patch_W   = (const float*)d_in[5];
  const float* patch_b   = (const float*)d_in[6];
  const float* pos_emb   = (const float*)d_in[7];
  const float* enc_qkv_W = (const float*)d_in[8];
  const float* enc_qkv_b = (const float*)d_in[9];
  const float* enc_out_W = (const float*)d_in[10];
  const float* enc_out_b = (const float*)d_in[11];
  const float* enc_ln1_s = (const float*)d_in[12];
  const float* enc_ln1_b = (const float*)d_in[13];
  const float* enc_ff1_W = (const float*)d_in[14];
  const float* enc_ff1_b = (const float*)d_in[15];
  const float* enc_ff2_W = (const float*)d_in[16];
  const float* enc_ff2_b = (const float*)d_in[17];
  const float* enc_ln2_s = (const float*)d_in[18];
  const float* enc_ln2_b = (const float*)d_in[19];
  const float* dsa_qkv_W = (const float*)d_in[20];
  const float* dsa_qkv_b = (const float*)d_in[21];
  const float* dsa_out_W = (const float*)d_in[22];
  const float* dsa_out_b = (const float*)d_in[23];
  const float* dca_qkv_W = (const float*)d_in[24];
  const float* dca_qkv_b = (const float*)d_in[25];
  const float* dca_out_W = (const float*)d_in[26];
  const float* dca_out_b = (const float*)d_in[27];
  const float* dec_ln1_s = (const float*)d_in[28];
  const float* dec_ln1_b = (const float*)d_in[29];
  const float* dec_ln2_s = (const float*)d_in[30];
  const float* dec_ln2_b = (const float*)d_in[31];
  const float* dec_ln3_s = (const float*)d_in[32];
  const float* dec_ln3_b = (const float*)d_in[33];
  const float* dec_ff1_W = (const float*)d_in[34];
  const float* dec_ff1_b = (const float*)d_in[35];
  const float* dec_ff2_W = (const float*)d_in[36];
  const float* dec_ff2_b = (const float*)d_in[37];
  const float* dec_nrm_s = (const float*)d_in[38];
  const float* dec_nrm_b = (const float*)d_in[39];
  const float* sk1_W     = (const float*)d_in[40];
  const float* sk1_b     = (const float*)d_in[41];
  const float* sk2_W     = (const float*)d_in[42];
  const float* sk2_b     = (const float*)d_in[43];
  const float* sk3_W     = (const float*)d_in[44];
  const float* sk3_b     = (const float*)d_in[45];
  const float* act_W     = (const float*)d_in[46];
  const float* act_b     = (const float*)d_in[47];
  const float* l1_W      = (const float*)d_in[48];
  const float* l1_b      = (const float*)d_in[49];
  const float* l2_W      = (const float*)d_in[50];
  const float* l2_b      = (const float*)d_in[51];
  (void)in_sizes; (void)n_in; (void)out_size; (void)ws_size;

  float* W = (float*)d_ws;
  size_t off = 0;
  auto alloc = [&](size_t n) { size_t r = off; off += (n + 7) & ~(size_t)7; return r; };
  const size_t MT = 50176;  // NP * BS
  float* src   = W + alloc(MT * 128);
  float* qkvE  = W + alloc(MT * 384);
  float* sc    = W + alloc(256UL * 208 * 208);
  float* ao    = W + alloc(MT * 128);
  float* t2    = W + alloc(MT * 128);
  float* ffb   = W + alloc(MT * 64);
  float* skp   = W + alloc(256UL * 224);
  float* s1    = W + alloc(256UL * 256);
  float* s2b   = W + alloc(256UL * 256);
  float* s3    = W + alloc(256UL * 100);
  float* feats = W + alloc(3840UL * 128);
  float* tgt   = W + alloc(3840UL * 128);
  float* tgt2  = W + alloc(3840UL * 128);
  float* qkvD  = W + alloc(3840UL * 384);
  float* qc    = W + alloc(3840UL * 128);
  float* h1    = W + alloc(3840UL * 128);
  float* ffd   = W + alloc(3840UL * 64);
  float* kc    = qkvE;             // cross-attn K (qkvE dead by then)
  float* vc    = qkvE + MT * 128;  // cross-attn V

  _Float16* H = (_Float16*)(W + off);
  size_t hoff = 0;
  auto prepack = [&](const float* Bsrc, int ldb, int K, int N) -> _Float16* {
    int Kt = (K + 31) / 32;
    int Nt = (((N + 15) / 16) + 3) & ~3;
    size_t total = (size_t)Nt * Kt * 32;
    _Float16* dst = H + hoff;
    hoff += total * 16;
    prepack_weight<<<dim3((unsigned)((total + 127) / 128)), 128, 0, stream>>>(
        Bsrc, ldb, K, N, dst, Kt, Nt);
    return dst;
  };

  _Float16* p_patch  = prepack(patch_W, 128, 768, 128);
  _Float16* p_eqkv   = prepack(enc_qkv_W, 384, 128, 384);
  _Float16* p_eout   = prepack(enc_out_W, 128, 128, 128);
  _Float16* p_eff1   = prepack(enc_ff1_W, 64, 128, 64);
  _Float16* p_eff2   = prepack(enc_ff2_W, 128, 64, 128);
  _Float16* p_sk1    = prepack(sk1_W, 256, 200, 256);
  _Float16* p_sk2    = prepack(sk2_W, 256, 256, 256);
  _Float16* p_sk3    = prepack(sk3_W, 100, 256, 100);
  _Float16* p_act    = prepack(act_W, 128, 118, 128);
  _Float16* p_dsaqkv = prepack(dsa_qkv_W, 384, 128, 384);
  _Float16* p_dsaout = prepack(dsa_out_W, 128, 128, 128);
  _Float16* p_dcaqkv = prepack(dca_qkv_W, 384, 128, 384);
  _Float16* p_dcaout = prepack(dca_out_W, 128, 128, 128);
  _Float16* p_dff1   = prepack(dec_ff1_W, 64, 128, 64);
  _Float16* p_dff2   = prepack(dec_ff2_W, 128, 64, 128);
  _Float16* p_l1     = prepack(l1_W, 128, 128, 128);
  const size_t dslice = (size_t)8 * 4 * 512;  // 8 N-tiles * Kt(4) * 512 halves
  _Float16* p_dcaq = p_dcaqkv;
  _Float16* p_dcak = p_dcaqkv + dslice;
  _Float16* p_dcav = p_dcaqkv + 2 * dslice;

  const dim3 blk(128);
  const float scale = 0.17677669529663687f;  // 1/sqrt(32)

  // ---- Encoder ----
  patch_embed_kernel<<<dim3(784, 2), blk, 0, stream>>>(image, p_patch, patch_b,
                                                       pos_emb, src);
  gemm_wmma<<<dim3(784, 6), blk, 0, stream>>>(src, 128, p_eqkv, 4, enc_qkv_b,
                                              qkvE, 384, 50176, 384, 0);
  for (int c = 0; c < 4; ++c) {
    attn_scores<<<dim3(13, 13, 256), 32, 0, stream>>>(
        qkvE, 384, 0, qkvE, 384, 128, sc, 208, 208, 196, 196, c * 256, scale);
    softmax_rows<<<dim3(12544), blk, 0, stream>>>(sc, 208, 208, 196, 196);
    attn_out<<<dim3(13, 2, 256), 32, 0, stream>>>(
        sc, 208, 208, qkvE, 384, 256, ao, 128, 0, 196, 196, c * 256);
  }
  gemm_wmma<<<dim3(784, 2), blk, 0, stream>>>(ao, 128, p_eout, 4, enc_out_b, t2,
                                              128, 50176, 128, 0);
  layernorm_k<<<dim3(12544), blk, 0, stream>>>(src, t2, enc_ln1_s, enc_ln1_b,
                                               src, 50176);
  gemm_wmma<<<dim3(784, 1), blk, 0, stream>>>(src, 128, p_eff1, 4, enc_ff1_b,
                                              ffb, 64, 50176, 64, 1);
  gemm_wmma<<<dim3(784, 2), blk, 0, stream>>>(ffb, 64, p_eff2, 2, enc_ff2_b, t2,
                                              128, 50176, 128, 0);
  layernorm_k<<<dim3(12544), blk, 0, stream>>>(src, t2, enc_ln2_s, enc_ln2_b,
                                               src, 50176);

  // ---- Sketch MLP ----
  pad_sk<<<dim3((256 * 224 + 127) / 128), blk, 0, stream>>>(sk, skp);
  gemm_wmma<<<dim3(4, 4), blk, 0, stream>>>(skp, 224, p_sk1, 7, sk1_b, s1, 256,
                                            256, 256, 1);
  gemm_wmma<<<dim3(4, 4), blk, 0, stream>>>(s1, 256, p_sk2, 8, sk2_b, s2b, 256,
                                            256, 256, 1);
  gemm_wmma<<<dim3(4, 2), blk, 0, stream>>>(s2b, 256, p_sk3, 8, sk3_b, s3, 100,
                                            256, 100, 0);

  // ---- Action tokens ----
  build_feats<<<dim3((3840 * 128 + 127) / 128), blk, 0, stream>>>(
      angle, pos_x, pos_y, s3, feats);
  gemm_wmma<<<dim3(60, 2), blk, 0, stream>>>(feats, 128, p_act, 4, act_b, tgt,
                                             128, 3840, 128, 0);

  // ---- Decoder self-attention ----
  gemm_wmma<<<dim3(60, 6), blk, 0, stream>>>(tgt, 128, p_dsaqkv, 4, dsa_qkv_b,
                                             qkvD, 384, 3840, 384, 0);
  attn_scores<<<dim3(1, 1, 1024), 32, 0, stream>>>(qkvD, 384, 0, qkvD, 384, 128,
                                                   sc, 16, 16, 15, 15, 0, scale);
  softmax_rows<<<dim3(3840), blk, 0, stream>>>(sc, 16, 16, 15, 15);
  attn_out<<<dim3(1, 2, 1024), 32, 0, stream>>>(sc, 16, 16, qkvD, 384, 256,
                                                tgt2, 128, 0, 15, 15, 0);
  gemm_wmma<<<dim3(60, 2), blk, 0, stream>>>(tgt2, 128, p_dsaout, 4, dsa_out_b,
                                             t2, 128, 3840, 128, 0);
  layernorm_k<<<dim3(960), blk, 0, stream>>>(tgt, t2, dec_ln1_s, dec_ln1_b, tgt,
                                             3840);

  // ---- Decoder cross-attention ----
  gemm_wmma<<<dim3(60, 2), blk, 0, stream>>>(tgt, 128, p_dcaq, 4, dca_qkv_b, qc,
                                             128, 3840, 128, 0);
  gemm_wmma<<<dim3(784, 2), blk, 0, stream>>>(src, 128, p_dcak, 4,
                                              dca_qkv_b + 128, kc, 128, 50176,
                                              128, 0);
  gemm_wmma<<<dim3(784, 2), blk, 0, stream>>>(src, 128, p_dcav, 4,
                                              dca_qkv_b + 256, vc, 128, 50176,
                                              128, 0);
  attn_scores<<<dim3(1, 13, 1024), 32, 0, stream>>>(qc, 128, 0, kc, 128, 0, sc,
                                                    208, 16, 15, 196, 0, scale);
  softmax_rows<<<dim3(3840), blk, 0, stream>>>(sc, 208, 16, 15, 196);
  attn_out<<<dim3(1, 2, 1024), 32, 0, stream>>>(sc, 208, 16, vc, 128, 0, tgt2,
                                                128, 0, 15, 196, 0);
  gemm_wmma<<<dim3(60, 2), blk, 0, stream>>>(tgt2, 128, p_dcaout, 4, dca_out_b,
                                             t2, 128, 3840, 128, 0);
  layernorm_k<<<dim3(960), blk, 0, stream>>>(tgt, t2, dec_ln2_s, dec_ln2_b, tgt,
                                             3840);

  // ---- Decoder FF + norms ----
  gemm_wmma<<<dim3(60, 1), blk, 0, stream>>>(tgt, 128, p_dff1, 4, dec_ff1_b,
                                             ffd, 64, 3840, 64, 1);
  gemm_wmma<<<dim3(60, 2), blk, 0, stream>>>(ffd, 64, p_dff2, 2, dec_ff2_b, t2,
                                             128, 3840, 128, 0);
  layernorm_k<<<dim3(960), blk, 0, stream>>>(tgt, t2, dec_ln3_s, dec_ln3_b, tgt,
                                             3840);
  layernorm_k<<<dim3(960), blk, 0, stream>>>(tgt, nullptr, dec_nrm_s, dec_nrm_b,
                                             tgt, 3840);

  // ---- Q head ----
  gemm_wmma<<<dim3(60, 2), blk, 0, stream>>>(tgt, 128, p_l1, 4, l1_b, h1, 128,
                                             3840, 128, 1);
  head_final<<<dim3(960), blk, 0, stream>>>(h1, l2_W, l2_b, (float*)d_out);
}
